// TestTimeAdaptationSystem_30906584662257
// MI455X (gfx1250) — matrix operand order repaired and checked by reading, not compiled, not run on hardware
//
#include <hip/hip_runtime.h>
#include <hip/hip_bf16.h>
#include <math.h>

#define B_ 1024
#define M_ 16384
#define D_ 256
#define A_ 264
#define H_ 8
#define DH_ 33

typedef __attribute__((ext_vector_type(2)))  float  v2f;
typedef __attribute__((ext_vector_type(8)))  float  v8f;
typedef __attribute__((ext_vector_type(16))) __bf16 v16bf;

union BFrag { v16bf bf; unsigned int u[8]; };

__device__ __forceinline__ v8f wmma_f32x4(v2f a, v2f b, v8f c) {
  return __builtin_amdgcn_wmma_f32_16x16x4_f32(false, a, false, b, (short)0, c, false, false);
}
__device__ __forceinline__ v8f wmma_bf16(v16bf a, v16bf b, v8f c) {
  return __builtin_amdgcn_wmma_f32_16x16x32_bf16(false, a, false, b, (short)0, c, false, false);
}
__device__ __forceinline__ unsigned short f2bf(float f) {
  unsigned int u = __builtin_bit_cast(unsigned int, f);
  u += 0x7fffu + ((u >> 16) & 1u);
  return (unsigned short)(u >> 16);
}
// load a 16-bf16 fragment as two 16B groups at ushort offsets g0 / g1
__device__ __forceinline__ void ld_frag(BFrag& f, const unsigned short* base, int g0, int g1) {
  uint4 a = *(const uint4*)(base + g0);
  uint4 b = *(const uint4*)(base + g1);
  f.u[0] = a.x; f.u[1] = a.y; f.u[2] = a.z; f.u[3] = a.w;
  f.u[4] = b.x; f.u[5] = b.y; f.u[6] = b.z; f.u[7] = b.w;
}
// async global -> LDS 16B copy (CDNA5 ISA), per-lane addresses
__device__ __forceinline__ void async_g2l_b128(const void* lds_dst, const void* gptr) {
  unsigned lds_off = (unsigned)(size_t)lds_dst;
  unsigned long long ga = (unsigned long long)(size_t)gptr;
  asm volatile("global_load_async_to_lds_b128 %0, %1, off" :: "v"(lds_off), "v"(ga) : "memory");
}
__device__ __forceinline__ void wait_async0() {
  asm volatile("s_wait_asynccnt 0x0" ::: "memory");
}

// ---------------------------------------------------------------- zero ws
__global__ void zero_kernel(unsigned int* __restrict__ p, size_t n) {
  size_t i = (size_t)blockIdx.x * blockDim.x + threadIdx.x;
  size_t st = (size_t)gridDim.x * blockDim.x;
  for (; i < n; i += st) p[i] = 0u;
}

// ---------------------------------------------------------------- row squared-norms / norms
__global__ __launch_bounds__(256) void sqnorm_kernel(const float* __restrict__ src,
                                                     float* __restrict__ dst,
                                                     int rows, int cols, int doSqrt) {
  int w = (blockIdx.x * (blockDim.x >> 5)) + (threadIdx.x >> 5);
  int lane = threadIdx.x & 31;
  if (w >= rows) return;
  const float* r = src + (size_t)w * cols;
  float s = 0.f;
  for (int c = lane; c < cols; c += 32) { float x = r[c]; s += x * x; }
#pragma unroll
  for (int m = 16; m >= 1; m >>= 1) s += __shfl_xor(s, m);
  if (lane == 0) dst[w] = doSqrt ? sqrtf(s) : s;
}

// ---------------------------------------------------------------- surprise: min-distance via fp32 WMMA
__global__ __launch_bounds__(128) void surprise_kernel(const float* __restrict__ F,
                                                       const float* __restrict__ Km,
                                                       const float* __restrict__ f2,
                                                       const float* __restrict__ m2,
                                                       const float* __restrict__ gmag,
                                                       float* __restrict__ surprise) {
  const int FS = D_ + 4;   // 260: 16B-aligned rows, 4-bank rotation
  __shared__ __align__(16) float Ftile[16 * FS];
  __shared__ float smin[4 * 16];
  int tid = threadIdx.x;
  int rowBase = blockIdx.x * 16;
  for (int i = tid * 4; i < 16 * D_; i += 512) {
    int r = i >> 8, c = i & 255;
    async_g2l_b128(&Ftile[r * FS + c], F + (size_t)(rowBase + r) * D_ + c);
  }
  wait_async0();
  __syncthreads();
  int w = tid >> 5, lane = tid & 31, half = lane >> 4, lr = lane & 15;
  float qf2[8];
#pragma unroll
  for (int j = 0; j < 8; ++j) qf2[j] = f2[rowBase + j + 8 * half];
  v8f dmin;
#pragma unroll
  for (int j = 0; j < 8; ++j) dmin[j] = 3.4e38f;

  const float* fr = Ftile + lr * FS;
  for (int t = w; t < M_ / 16; t += 4) {
    int key = t * 16 + lr;
    const float* kr = Km + (size_t)key * D_;
    v8f acc = {};
#pragma unroll 8
    for (int k0 = 0; k0 < D_; k0 += 4) {
      int kk = k0 + 2 * half;
      v2f a; a.x = fr[kk]; a.y = fr[kk + 1];
      v2f b; b.x = kr[kk]; b.y = kr[kk + 1];
      acc = wmma_f32x4(a, b, acc);
    }
    float km2 = m2[key];
#pragma unroll
    for (int j = 0; j < 8; ++j) {
      float d2 = qf2[j] + km2 - 2.0f * acc[j];
      dmin[j] = fminf(dmin[j], d2);
    }
  }
#pragma unroll
  for (int j = 0; j < 8; ++j) {
    float v = dmin[j];
#pragma unroll
    for (int m = 1; m <= 8; m <<= 1) v = fminf(v, __shfl_xor(v, m));
    dmin[j] = v;
  }
  if (lr == 0)
    for (int j = 0; j < 8; ++j) smin[w * 16 + j + 8 * half] = dmin[j];
  __syncthreads();
  if (tid < 16) {
    float v = fminf(fminf(smin[tid], smin[16 + tid]), fminf(smin[32 + tid], smin[48 + tid]));
    surprise[rowBase + tid] = gmag[rowBase + tid] * sqrtf(fmaxf(v, 0.0f));
  }
}

// ---------------------------------------------------------------- fused two-stage projection
// stage1: T1 = X @ W1^T + b1  (N1 = 264, in LDS), stage2: out = T1 @ W2^T + b2
// MODE 0: bf16 packed [H][rows][64]; MODE 1: bf16 packed transposed [H][64][mtot]; MODE 2: fp32 [rows][N2]
template <int K1, int N2, int MODE>
__global__ __launch_bounds__(128) void proj2_kernel(const float* __restrict__ X, int rows,
                                                    const float* __restrict__ W1,
                                                    const float* __restrict__ b1,
                                                    const float* __restrict__ W2,
                                                    const float* __restrict__ b2,
                                                    void* __restrict__ outp, int mtot) {
  const int XS = K1 + 4;
  const int TS = A_ + 2;
  __shared__ __align__(16) float Xs[16 * XS];
  __shared__ float T1[16 * TS];
  int tid = threadIdx.x;
  int rowBase = blockIdx.x * 16;
  for (int i = tid * 4; i < 16 * K1; i += 512) {
    int r = i / K1, c = i - r * K1;
    async_g2l_b128(&Xs[r * XS + c], X + (size_t)(rowBase + r) * K1 + c);
  }
  wait_async0();
  __syncthreads();
  int w = tid >> 5, lane = tid & 31, half = lane >> 4, lr = lane & 15;

  for (int t = w; t < 17; t += 4) {
    int col = t * 16 + lr;
    int colc = col < A_ ? col : 0;
    float msk = col < A_ ? 1.0f : 0.0f;
    const float* wr = W1 + (size_t)colc * K1;
    const float* xr = Xs + lr * XS;
    v8f acc = {};
#pragma unroll 8
    for (int k0 = 0; k0 < K1; k0 += 4) {
      int kk = k0 + 2 * half;
      v2f a; a.x = xr[kk];       a.y = xr[kk + 1];
      v2f b; b.x = wr[kk] * msk; b.y = wr[kk + 1] * msk;
      acc = wmma_f32x4(a, b, acc);
    }
    if (col < A_) {
      float bias = b1[col];
#pragma unroll
      for (int j = 0; j < 8; ++j) T1[(j + 8 * half) * TS + col] = acc[j] + bias;
    }
  }
  __syncthreads();

  const int NT2 = (N2 + 15) / 16;
  for (int t = w; t < NT2; t += 4) {
    int col = t * 16 + lr;
    int colc = col < N2 ? col : 0;
    float msk = col < N2 ? 1.0f : 0.0f;
    const float* wr = W2 + (size_t)colc * A_;
    const float* tr = T1 + lr * TS;
    v8f acc = {};
#pragma unroll 6
    for (int k0 = 0; k0 < A_; k0 += 4) {
      int kk = k0 + 2 * half;
      v2f a; a.x = tr[kk];       a.y = tr[kk + 1];
      v2f b; b.x = wr[kk] * msk; b.y = wr[kk + 1] * msk;
      acc = wmma_f32x4(a, b, acc);
    }
    if (col < N2) {
      float bias = b2[col];
      if (MODE == 2) {
        float* out = (float*)outp;
#pragma unroll
        for (int j = 0; j < 8; ++j) {
          int m = rowBase + j + 8 * half;
          out[(size_t)m * N2 + col] = acc[j] + bias;
        }
      } else {
        unsigned short* out = (unsigned short*)outp;
        unsigned int h = (unsigned int)col / DH_;
        unsigned int d = (unsigned int)col - h * DH_;
#pragma unroll
        for (int j = 0; j < 8; ++j) {
          int m = rowBase + j + 8 * half;
          unsigned short v = f2bf(acc[j] + bias);
          if (MODE == 0) out[((size_t)h * rows + m) * 64 + d] = v;
          else           out[((size_t)h * 64 + d) * mtot + m] = v;
        }
      }
    }
  }
}

// ---------------------------------------------------------------- flash attention, bf16 WMMA
// S^T formulation: A = K-tile (rows = keys), B = Q-tile (cols = queries)
// -> each lane owns ONE query column; softmax state is per-lane scalar.
__global__ __launch_bounds__(256) void attention_kernel(const unsigned short* __restrict__ qp,
                                                        const unsigned short* __restrict__ kp,
                                                        const unsigned short* __restrict__ vpT,
                                                        float* __restrict__ ctx) {
  const int PS = 40;  // ushorts per P row: 80B, 16B aligned, 20-bank rotation
  __shared__ __align__(16) unsigned short Ps[8 * 16 * PS];
  int tid = threadIdx.x;
  int h = tid >> 5;  // one wave == one head
  int lane = tid & 31, half = lane >> 4, lr = lane & 15;
  int qBase = blockIdx.x * 16;
  unsigned short* myPs = Ps + h * 16 * PS;
  const int gA0 = 8 * half,  gA1 = 16 + 8 * half;   // 16-bit A 16x32 group offsets
  const int gB0 = 8 * half,  gB1 = 16 + 8 * half;   // 16-bit B 32x16 group offsets

  // Q as B-operand fragments (K(d) = 0..31 and 32..63), preloaded once
  BFrag qlo, qhi;
  {
    const unsigned short* qrow = qp + ((size_t)h * B_ + qBase + lr) * 64;
    ld_frag(qlo, qrow, gB0, gB1);
    ld_frag(qhi, qrow + 32, gB0, gB1);
  }

  const float scale = 0.174077656f;  // 1/sqrt(33)
  float runmax = -3.4e38f, runsum = 0.f;   // per-lane: this lane's query = lr
  v8f cacc[3];
#pragma unroll
  for (int t = 0; t < 3; ++t) cacc[t] = (v8f){};

  const unsigned short* kbh = kp + (size_t)h * M_ * 64;
  const unsigned short* vbh = vpT + (size_t)h * 64 * M_;
  unsigned short* prow = myPs + lr * PS;

  for (int m0 = 0; m0 < M_; m0 += 32) {
    if (m0 + 32 < M_) {
      __builtin_prefetch(kbh + (size_t)(m0 + 32 + lr) * 64, 0, 1);
      __builtin_prefetch(vbh + (size_t)lr * M_ + m0 + 32, 0, 1);
    }
    // ---- S^T: two 16(key)x16(query) tiles, K = 64 (padded head dim)
    v8f st0 = {}, st1 = {};
    {
      const unsigned short* kr0 = kbh + (size_t)(m0 + lr) * 64;
      const unsigned short* kr1 = kbh + (size_t)(m0 + 16 + lr) * 64;
      BFrag a0l, a0h, a1l, a1h;
      ld_frag(a0l, kr0, gA0, gA1);
      ld_frag(a0h, kr0 + 32, gA0, gA1);
      ld_frag(a1l, kr1, gA0, gA1);
      ld_frag(a1h, kr1 + 32, gA0, gA1);
      st0 = wmma_bf16(a0l.bf, qlo.bf, st0);
      st0 = wmma_bf16(a0h.bf, qhi.bf, st0);
      st1 = wmma_bf16(a1l.bf, qlo.bf, st1);
      st1 = wmma_bf16(a1h.bf, qhi.bf, st1);
    }
    // ---- online softmax for this lane's query over its 16 key-scores
    float c[16];
#pragma unroll
    for (int j = 0; j < 8; ++j) { c[j] = st0[j] * scale; c[8 + j] = st1[j] * scale; }
    float mloc = c[0];
#pragma unroll
    for (int i = 1; i < 16; ++i) mloc = fmaxf(mloc, c[i]);
    mloc = fmaxf(mloc, __shfl_xor(mloc, 16));       // combine the two key halves
    float mnew = fmaxf(runmax, mloc);
    float corr = __expf(runmax - mnew);
    runmax = mnew;
    float p[16], ps = 0.f;
#pragma unroll
    for (int i = 0; i < 16; ++i) { p[i] = __expf(c[i] - mnew); ps += p[i]; }
    ps += __shfl_xor(ps, 16);
    runsum = runsum * corr + ps;
    // ---- pack P (bf16) and store: row = query lr, cols = this lane's 2x8 keys
    uint4 pk0, pk1;
    pk0.x = f2bf(p[0])  | ((unsigned)f2bf(p[1])  << 16);
    pk0.y = f2bf(p[2])  | ((unsigned)f2bf(p[3])  << 16);
    pk0.z = f2bf(p[4])  | ((unsigned)f2bf(p[5])  << 16);
    pk0.w = f2bf(p[6])  | ((unsigned)f2bf(p[7])  << 16);
    pk1.x = f2bf(p[8])  | ((unsigned)f2bf(p[9])  << 16);
    pk1.y = f2bf(p[10]) | ((unsigned)f2bf(p[11]) << 16);
    pk1.z = f2bf(p[12]) | ((unsigned)f2bf(p[13]) << 16);
    pk1.w = f2bf(p[14]) | ((unsigned)f2bf(p[15]) << 16);
    *(uint4*)(prow + 8 * half)      = pk0;  // keys 8*half .. +7
    *(uint4*)(prow + 16 + 8 * half) = pk1;  // keys 16+8*half .. +7
    // ---- rescale ctx accumulators (corr broadcast from the lane owning that query)
#pragma unroll
    for (int j = 0; j < 8; ++j) {
      float cj = __shfl(corr, j + 8 * half);
      cacc[0][j] *= cj; cacc[1][j] *= cj; cacc[2][j] *= cj;
    }
    asm volatile("s_wait_dscnt 0x0" ::: "memory");  // LDS RAW within wave
    // ---- ctx += P @ V  (A = P 16x32 from LDS, K = 32 keys, 3 N-tiles cover DH=33)
    BFrag ap;
    ld_frag(ap, prow, gA0, gA1);
#pragma unroll
    for (int t = 0; t < 3; ++t) {
      BFrag bv;
      const unsigned short* vr = vbh + (size_t)(t * 16 + lr) * M_ + m0;
      ld_frag(bv, vr, gB0, gB1);
      cacc[t] = wmma_bf16(ap.bf, bv.bf, cacc[t]);
    }
  }
  float inv = 1.0f / runsum;
#pragma unroll
  for (int t = 0; t < 3; ++t) {
    int d = t * 16 + lr;
    if (d < DH_) {
#pragma unroll
      for (int j = 0; j < 8; ++j) {
        float rj = __shfl(inv, j + 8 * half);
        int b = qBase + j + 8 * half;
        ctx[(size_t)b * A_ + h * DH_ + d] = cacc[t][j] * rj;
      }
    }
  }
}

// ---------------------------------------------------------------- workspace layout
constexpr size_t OFF_F2   = 0;
constexpr size_t OFF_M2   = 4096;
constexpr size_t OFF_GMAG = OFF_M2 + (size_t)M_ * 4;
constexpr size_t OFF_CTX  = OFF_GMAG + (size_t)B_ * 4;
constexpr size_t OFF_QP   = OFF_CTX + (size_t)B_ * A_ * 4;
constexpr size_t OFF_KP   = OFF_QP + (size_t)H_ * B_ * 64 * 2;
constexpr size_t OFF_VPT  = OFF_KP + (size_t)H_ * M_ * 64 * 2;
constexpr size_t WS_END   = OFF_VPT + (size_t)H_ * 64 * M_ * 2;

extern "C" void kernel_launch(void* const* d_in, const int* in_sizes, int n_in,
                              void* d_out, int out_size, void* d_ws, size_t ws_size,
                              hipStream_t stream) {
  (void)in_sizes; (void)n_in; (void)out_size; (void)ws_size;
  const float* features      = (const float*)d_in[0];
  const float* gradients     = (const float*)d_in[1];
  const float* memory_keys   = (const float*)d_in[2];
  const float* memory_values = (const float*)d_in[3];
  const float* Wf  = (const float*)d_in[4];
  const float* bf  = (const float*)d_in[5];
  const float* Wq  = (const float*)d_in[6];
  const float* Wk  = (const float*)d_in[7];
  const float* Wv  = (const float*)d_in[8];
  const float* bq  = (const float*)d_in[9];
  const float* bk  = (const float*)d_in[10];
  const float* bv  = (const float*)d_in[11];
  const float* Wo  = (const float*)d_in[12];
  const float* bo  = (const float*)d_in[13];
  const float* Wout = (const float*)d_in[14];
  const float* bout = (const float*)d_in[15];

  char* ws = (char*)d_ws;
  float* f2   = (float*)(ws + OFF_F2);
  float* m2   = (float*)(ws + OFF_M2);
  float* gmag = (float*)(ws + OFF_GMAG);
  float* ctx  = (float*)(ws + OFF_CTX);
  unsigned short* qp  = (unsigned short*)(ws + OFF_QP);
  unsigned short* kp  = (unsigned short*)(ws + OFF_KP);
  unsigned short* vpT = (unsigned short*)(ws + OFF_VPT);

  float* retrieved = (float*)d_out;                    // [B, D]
  float* surprise  = (float*)d_out + (size_t)B_ * D_;  // [B]

  // zero packed bf16 buffers (pad lanes must be exact zero for S-GEMM K-padding)
  size_t zwords = (WS_END - OFF_QP) / 4;
  zero_kernel<<<2048, 256, 0, stream>>>((unsigned int*)(ws + OFF_QP), zwords);

  sqnorm_kernel<<<(B_ + 7) / 8, 256, 0, stream>>>(features,    f2,   B_, D_, 0);
  sqnorm_kernel<<<(M_ + 7) / 8, 256, 0, stream>>>(memory_keys, m2,   M_, D_, 0);
  sqnorm_kernel<<<(B_ + 7) / 8, 256, 0, stream>>>(gradients,   gmag, B_, D_, 1);

  surprise_kernel<<<B_ / 16, 128, 0, stream>>>(features, memory_keys, f2, m2, gmag, surprise);

  proj2_kernel<D_, A_, 0><<<B_ / 16, 128, 0, stream>>>(features,      B_, Wf, bf, Wq, bq, (void*)qp,  B_);
  proj2_kernel<D_, A_, 0><<<M_ / 16, 128, 0, stream>>>(memory_keys,   M_, Wf, bf, Wk, bk, (void*)kp,  M_);
  proj2_kernel<D_, A_, 1><<<M_ / 16, 128, 0, stream>>>(memory_values, M_, Wf, bf, Wv, bv, (void*)vpT, M_);

  attention_kernel<<<B_ / 16, 256, 0, stream>>>(qp, kp, vpT, ctx);

  proj2_kernel<A_, D_, 2><<<B_ / 16, 128, 0, stream>>>(ctx, B_, Wo, bo, Wout, bout, (void*)retrieved, 0);
}
